// IterativeConvLSTM2D_90391881711775
// MI455X (gfx1250) — matrix operand order repaired and checked by reading, not compiled, not run on hardware
//
#include <hip/hip_runtime.h>

// ---------------------------------------------------------------------------
// Types matching CDNA5 WMMA builtin signatures
// ---------------------------------------------------------------------------
typedef __attribute__((ext_vector_type(16))) __bf16        v16bf;
typedef __attribute__((ext_vector_type(8)))  float         v8f;
typedef __attribute__((ext_vector_type(8)))  unsigned int  v8u;
typedef __attribute__((ext_vector_type(4)))  int           v4i;

#define B_   16
#define T_   16
#define H_   64
#define W_   64
#define F_   64
#define N_   256        // 4F
#define HW_  (H_*W_)
#define HWF_ (H_*W_*F_)

// packed-weight sizes (u32 units)
#define WPACK_PER_T (9*2*16*32*8)      // 73728 u32 = 288KB per step
#define WPACK_TOTAL (T_*WPACK_PER_T)   // 1179648 u32 = 4.71MB

// CDNA5 async global->LDS copy path (direct LDS write, ASYNCcnt tracked).
#if defined(__has_builtin)
#if __has_builtin(__builtin_amdgcn_global_load_async_to_lds_b128) && \
    __has_builtin(__builtin_amdgcn_s_wait_asynccnt)
#define USE_ASYNC_LDS 1
#endif
#endif

#if defined(USE_ASYNC_LDS)
typedef __attribute__((address_space(1))) v4i* gv4i_p;   // global int4*
typedef __attribute__((address_space(3))) v4i* lv4i_p;   // LDS int4*
#endif

__device__ __forceinline__ unsigned int f2bf1(float f) {
    unsigned int u = __builtin_bit_cast(unsigned int, f);
    u += 0x7FFFu + ((u >> 16) & 1u);        // round-to-nearest-even
    return u >> 16;
}
__device__ __forceinline__ float hsig(float z) {
    return fminf(fmaxf(0.2f * z + 0.5f, 0.0f), 1.0f);
}

// ---------------------------------------------------------------------------
// Kernel 1: combine (W + U) for all T steps, convert to bf16, and pack into
// this code's contiguous-K v_wmma_f32_16x16x32_bf16 B fragment order (same as A):
//   B 32x16 (KxN): VGPR j, lanes 0-15 : K=2j,2j+1 (N=lane)
//                  VGPR j, lanes 16-31: K=16+2j,16+2j+1 (N=lane-16)
// packed index: ((((t*9+tap)*2+kt)*16+ntile)*32+lane)*8 + j   (u32)
// ---------------------------------------------------------------------------
__global__ __launch_bounds__(256) void convlstm_pack_weights(
    const float* __restrict__ kern,    // [T,3,3,F,4F] = [T,9,64,256]
    const float* __restrict__ rkern,   // same shape
    unsigned int* __restrict__ wpack)
{
    int id = blockIdx.x * 256 + threadIdx.x;
    if (id >= WPACK_TOTAL) return;

    int j     =  id        & 7;
    int lane  = (id >> 3)  & 31;
    int ntile = (id >> 8)  & 15;
    int kt    = (id >> 12) & 1;
    int rest  =  id >> 13;             // t*9 + tap
    int tap   = rest % 9;
    int t     = rest / 9;

    int grp  = lane >> 4;
    int nlo  = lane & 15;
    int n    = ntile * 16 + nlo;
    int k0   = grp * 16 + 2 * j;       // K within 32-slice (pair base)
    int kc   = kt * 32 + k0;           // channel within tap's 64

    long s0 = ((long)(t * 9 + tap) * F_ + kc) * N_ + n;
    float a0 = kern[s0]        + rkern[s0];
    float a1 = kern[s0 + N_]   + rkern[s0 + N_];     // kc+1
    wpack[id] = f2bf1(a0) | (f2bf1(a1) << 16);
}

// ---------------------------------------------------------------------------
// B K-slice staging: 16KB (4096 u32), 128 threads.
// Async path: GLOBAL_LOAD_ASYNC_TO_LDS_B128 (no VGPR bounce, ASYNCcnt).
// Fallback: register copy.
// ---------------------------------------------------------------------------
__device__ __forceinline__ void stage_b(const unsigned int* __restrict__ gsrc,
                                        unsigned int* lds_dst, int tid)
{
    const uint4* s = (const uint4*)gsrc;
    uint4*       d = (uint4*)lds_dst;
#if defined(USE_ASYNC_LDS)
#pragma unroll
    for (int r = 0; r < 8; ++r) {
        int i = tid + r * 128;
        __builtin_amdgcn_global_load_async_to_lds_b128(
            (gv4i_p)(s + i), (lv4i_p)(d + i), 0, 0);
    }
#else
#pragma unroll
    for (int r = 0; r < 8; ++r) {
        int i = tid + r * 128;
        d[i] = s[i];
    }
#endif
}

// ---------------------------------------------------------------------------
// Kernel 2: one ConvLSTM timestep.
//   grid = B*H blocks; block = 128 threads (4 waves).
//   Block handles one (b, y) image row: M = 64 x-positions, full N = 256.
//   Wave w: x in [16w, 16w+16); 16 accumulators (ntile 0..15).
//   18 K-slices (9 taps x 2), double-buffered B staging overlapped with WMMA.
// ---------------------------------------------------------------------------
__global__ __launch_bounds__(128) void convlstm_step(
    const float* __restrict__ h_src,   // h_{t-1}: h0 or d_out slice
    long long h_bstride,               // batch stride of h_src (elements)
    const float* __restrict__ c_src,   // c_{t-1}
    float* __restrict__ c_dst,         // c_t
    float* __restrict__ out,           // d_out base [B,T,H,W,F]
    const unsigned int* __restrict__ wpack_t, // this step's packed weights
    const float* __restrict__ bias_t,  // this step's bias [256]
    int t)
{
    __shared__ unsigned int lds_h[3 * 66 * 32];    // 3 rows x 66 x (64ch as bf16 pairs)
    __shared__ unsigned int lds_b[2][16 * 32 * 8]; // double-buffered 32x256 B K-slice

    const int blk  = blockIdx.x;       // 0..B*H-1
    const int b    = blk >> 6;
    const int y    = blk & 63;
    const int tid  = threadIdx.x;
    const int wv   = tid >> 5;         // wave 0..3
    const int lane = tid & 31;
    const int grp  = lane >> 4;
    const int mlo  = lane & 15;

    // ---- kick off staging of B slice 0 ----
    stage_b(wpack_t, lds_b[0], tid);

    // ---- stage h halo (3 rows x 66 x 64ch) as bf16 pairs ----
    for (int i = tid; i < 3 * 66 * 32; i += 128) {
        int cpair = i & 31;            // channel/2
        int rem   = i >> 5;            // r*66 + hx
        int hx    = rem % 66;
        int r     = rem / 66;
        int yy    = y + r - 1;
        int xg    = hx - 1;
        unsigned int val = 0u;
        if (yy >= 0 && yy < H_ && xg >= 0 && xg < W_) {
            long long base = (long long)b * h_bstride
                           + ((long long)(yy * W_ + xg)) * F_ + cpair * 2;
            val = f2bf1(h_src[base]) | (f2bf1(h_src[base + 1]) << 16);
        }
        lds_h[i] = val;
    }
#if defined(USE_ASYNC_LDS)
    __builtin_amdgcn_s_wait_asynccnt(0);
#endif
    __syncthreads();

    // ---- accumulators: full N=256 (16 tiles of 16) ----
    v8f acc[16];
#pragma unroll
    for (int i = 0; i < 16; ++i) {
#pragma unroll
        for (int e = 0; e < 8; ++e) acc[i][e] = 0.0f;
    }

    // ---- pipelined K loop: 18 slices (tap 0..8 x kt 0..1) ----
    for (int s = 0; s < 18; ++s) {
        const int tap = s >> 1;
        const int kt  = s & 1;
        const int dy  = tap / 3;
        const int dx  = tap % 3;
        const unsigned int* bbuf = lds_b[s & 1];

        // prefetch next K-slice into the other buffer (overlaps with WMMAs)
        if (s + 1 < 18)
            stage_b(wpack_t + (s + 1) * 4096, lds_b[(s + 1) & 1], tid);

        // A fragment from halo LDS (lane grp -> K+16,
        // vgpr j -> K = 2*j, pairs contiguous in channel)
        const int hx = 16 * wv + mlo + dx;   // halo x for A row m = mlo
        v8u au;
#pragma unroll
        for (int j = 0; j < 8; ++j) {
            int jj = 2 * j;
            int c  = kt * 32 + grp * 16 + jj;         // even channel
            au[j] = lds_h[(dy * 66 + hx) * 32 + (c >> 1)];
        }
        v16bf afrag = __builtin_bit_cast(v16bf, au);

#pragma unroll
        for (int nt = 0; nt < 16; ++nt) {
            const unsigned int* bp = &bbuf[(nt * 32 + lane) * 8];
            v8u bu;
#pragma unroll
            for (int j = 0; j < 8; ++j) bu[j] = bp[j];
            v16bf bfrag = __builtin_bit_cast(v16bf, bu);
            acc[nt] = __builtin_amdgcn_wmma_f32_16x16x32_bf16(
                false, afrag, false, bfrag, (short)0, acc[nt], false, false);
        }

#if defined(USE_ASYNC_LDS)
        __builtin_amdgcn_s_wait_asynccnt(0);
#endif
        __syncthreads();   // next slice staged + current fully consumed
    }

    // ---- gates + state update ----
    // D layout: vgpr r, lane L: M = r + 8*(L/16), N = L%16.
    // Gates i,f,c,o live in acc[nt0], acc[nt0+4], acc[nt0+8], acc[nt0+12].
    const int x_base = 16 * wv;
#pragma unroll
    for (int nt0 = 0; nt0 < 4; ++nt0) {
        const int cc = nt0 * 16 + mlo;             // feature channel 0..63
        const float bi = bias_t[cc];
        const float bf = bias_t[64 + cc];
        const float bc = bias_t[128 + cc];
        const float bo = bias_t[192 + cc];
#pragma unroll
        for (int r = 0; r < 8; ++r) {
            const int x = x_base + r + 8 * grp;
            const long long cidx = ((long long)b * HW_ + y * W_ + x) * F_ + cc;
            float zi = acc[nt0][r]      + bi;
            float zf = acc[nt0 + 4][r]  + bf;
            float zc = acc[nt0 + 8][r]  + bc;
            float zo = acc[nt0 + 12][r] + bo;
            float ig = hsig(zi);
            float fg = hsig(zf);
            float og = hsig(zo);
            float cn = fg * c_src[cidx] + ig * tanhf(zc);
            float hn = og * tanhf(cn);
            c_dst[cidx] = cn;
            out[(((long long)b * T_ + t) * HW_ + y * W_ + x) * F_ + cc] = hn;
        }
    }
}

// ---------------------------------------------------------------------------
// Host launch: pack weights once, then 16 dependent step kernels.
// Workspace: [packed weights 4.71MB][c ping 16.78MB][c pong 16.78MB]
// ---------------------------------------------------------------------------
extern "C" void kernel_launch(void* const* d_in, const int* in_sizes, int n_in,
                              void* d_out, int out_size, void* d_ws, size_t ws_size,
                              hipStream_t stream) {
    (void)in_sizes; (void)n_in; (void)out_size; (void)ws_size;
    // inputs: 0=inputs(unused), 1=h0, 2=c0, 3=kernels, 4=rec_kernels, 5=bias
    const float* h0    = (const float*)d_in[1];
    const float* c0    = (const float*)d_in[2];
    const float* kern  = (const float*)d_in[3];
    const float* rkern = (const float*)d_in[4];
    const float* bias  = (const float*)d_in[5];
    float*       out   = (float*)d_out;

    unsigned int* wpack = (unsigned int*)d_ws;
    float* cping = (float*)((char*)d_ws + (size_t)WPACK_TOTAL * 4);
    float* cpong = cping + (size_t)B_ * HWF_;

    convlstm_pack_weights<<<(WPACK_TOTAL + 255) / 256, 256, 0, stream>>>(
        kern, rkern, wpack);

    for (int t = 0; t < T_; ++t) {
        const float* h_src = (t == 0) ? h0 : (out + (long long)(t - 1) * HWF_);
        long long h_bs     = (t == 0) ? (long long)HWF_ : (long long)T_ * HWF_;
        const float* c_src = (t == 0) ? c0 : ((t & 1) ? cping : cpong);
        float*       c_dst = (t & 1) ? cpong : cping;
        convlstm_step<<<B_ * H_, 128, 0, stream>>>(
            h_src, h_bs, c_src, c_dst, out,
            wpack + (size_t)t * WPACK_PER_T, bias + (size_t)t * N_, t);
    }
}
